// ScGPTSelfAttention_26310969655646
// MI455X (gfx1250) — matrix operand order repaired and checked by reading, not compiled, not run on hardware
//
#include <hip/hip_runtime.h>

// ---------------- problem constants ----------------
#define BATCH 2
#define SEQ   2048
#define DM    1024
#define NH    16
#define HD    64
#define NQKV  (3 * DM)          // 3072
#define MROWS (BATCH * SEQ)     // 4096

// GEMM workgroup tiling
#define GT_M 128
#define GT_N 128
#define GT_K 32

// ---------------- WMMA vector types ----------------
typedef __attribute__((ext_vector_type(16))) __bf16 v16bf;
typedef __attribute__((ext_vector_type(8)))  float  v8f;

union BfTile { v16bf v; uint4 q[2]; };

__device__ __forceinline__ unsigned short f2bf(float f) {
    unsigned int u = __float_as_uint(f);
    u += 0x7FFFu + ((u >> 16) & 1u);           // round-to-nearest-even
    return (unsigned short)(u >> 16);
}

// A-operand loader: 16x32 bf16 tile, row-major source with leading dim `ld`.
// ISA layout: lanes 0-15 hold row m=lane, K = {0..7, 16..23};
//             lanes 16-31 hold row m=lane-16, K = {8..15, 24..31}.
__device__ __forceinline__ v16bf loadA(const unsigned short* p, int ld, int lane) {
    const int row = lane & 15;
    const int kb  = (lane >> 4) << 3;          // 0 or 8
    const unsigned short* base = p + (size_t)row * ld + kb;
    BfTile t;
    t.q[0] = *(const uint4*)(base);            // K = kb .. kb+7
    t.q[1] = *(const uint4*)(base + 16);       // K = 16+kb .. 16+kb+7
    return t.v;
}

// B-operand loader: 32x16 bf16 tile (K x N), column n of B = row n of source.
// ISA layout: lanes 0-15 hold col n=lane, K = {0..7, 16..23};
//             lanes 16-31 hold col n=lane-16, K = {8..15, 24..31}.
__device__ __forceinline__ v16bf loadB(const unsigned short* p, int ld, int lane) {
    const int col = lane & 15;
    const int kb  = (lane >> 4) << 3;          // 0 or 8
    const unsigned short* base = p + (size_t)col * ld + kb;
    BfTile t;
    t.q[0] = *(const uint4*)(base);            // K = kb .. kb+7
    t.q[1] = *(const uint4*)(base + 16);       // K = 16+kb .. 16+kb+7
    return t.v;
}

__device__ __forceinline__ v8f wmma_bf16(v16bf a, v16bf b, v8f c) {
    return __builtin_amdgcn_wmma_f32_16x16x32_bf16(false, a, false, b,
                                                   (short)0, c, false, false);
}

// Async DMA: global -> LDS, 16 bytes per lane, tracked by ASYNCcnt.
__device__ __forceinline__ void async_load_b128(unsigned lds_off,
                                                const unsigned short* gptr) {
    asm volatile("global_load_async_to_lds_b128 %0, %1, off"
                 :: "v"(lds_off), "v"(gptr)
                 : "memory");
}

// ---------------- f32 -> bf16 conversion ----------------
__global__ void cvt_bf16_kernel(const float* __restrict__ src,
                                unsigned short* __restrict__ dst, int n) {
    int i = blockIdx.x * blockDim.x + threadIdx.x;
    if (i < n) dst[i] = f2bf(src[i]);
}

// ---------------- LDS-staged bf16 GEMM: C = A * Bw^T + bias ----------------
// A: [M, Kd] bf16 row-major. Bw: [N, Kd] bf16 row-major.
// Workgroup (256 thr = 8 waves) computes a 128x128 tile; waves in a 4(M) x 2(N)
// grid each compute 32x64 (2x4 WMMA tiles). A/B K-panels (128x32 bf16, 8KB each)
// are double-buffered in LDS and filled with global_load_async_to_lds_b128; the
// next panel's DMA stays in flight while WMMA runs on the current one
// (s_wait_asynccnt 4). Final iteration peeled so the hot loop is branch-free.
// mode 0: write f32 to Cout[M, N].
// mode 1: QKV epilogue -> scatter bf16 into Qs/Ks (b,h,s,d; q scaled) and Vt (b,h,d,s).
__global__ __launch_bounds__(256)
void gemm_bf16_lds_kernel(const unsigned short* __restrict__ A,
                          const unsigned short* __restrict__ Bw,
                          const float* __restrict__ bias,
                          float* __restrict__ Cout,
                          unsigned short* __restrict__ Qs,
                          unsigned short* __restrict__ Ks,
                          unsigned short* __restrict__ Vt,
                          int N, int Kd, int mode) {
    __shared__ unsigned short smA[2][GT_M * GT_K];   // 2 x 8KB
    __shared__ unsigned short smB[2][GT_N * GT_K];   // 2 x 8KB

    const int tid  = threadIdx.x;
    const int lane = tid & 31;
    const int wave = tid >> 5;
    const int wr   = wave >> 1;                 // 0..3 : 32-row strip in tile
    const int wc   = wave & 1;                  // 0..1 : 64-col strip in tile
    const int mb   = blockIdx.y * GT_M;
    const int nb   = blockIdx.x * GT_N;

    // two b128 slots per panel per thread: 512 x 16B = 8KB panel
    const int r0 = tid >> 2,         s0 = tid & 3;           // slot 0
    const int r1 = (tid + 256) >> 2, s1 = (tid + 256) & 3;   // slot 1

    auto issue_panels = [&](int buf, int kk) {
        const unsigned baseA = (unsigned)(size_t)(&smA[buf][0]);
        const unsigned baseB = (unsigned)(size_t)(&smB[buf][0]);
        async_load_b128(baseA + (unsigned)(r0 * 64 + s0 * 16),
                        A  + (size_t)(mb + r0) * Kd + kk + s0 * 8);
        async_load_b128(baseA + (unsigned)(r1 * 64 + s1 * 16),
                        A  + (size_t)(mb + r1) * Kd + kk + s1 * 8);
        async_load_b128(baseB + (unsigned)(r0 * 64 + s0 * 16),
                        Bw + (size_t)(nb + r0) * Kd + kk + s0 * 8);
        async_load_b128(baseB + (unsigned)(r1 * 64 + s1 * 16),
                        Bw + (size_t)(nb + r1) * Kd + kk + s1 * 8);
    };

    v8f acc[2][4];
#pragma unroll
    for (int i = 0; i < 2; ++i)
#pragma unroll
        for (int j = 0; j < 4; ++j) acc[i][j] = (v8f){};

    // issue all 12 ds_load_b128 first, then the 8 WMMAs -> partial DScnt waits
    auto compute_step = [&](int buf) {
        const unsigned short* pA = &smA[buf][0];
        const unsigned short* pB = &smB[buf][0];
        v16bf a0 = loadA(pA + (wr * 32 +  0) * GT_K, GT_K, lane);
        v16bf a1 = loadA(pA + (wr * 32 + 16) * GT_K, GT_K, lane);
        v16bf b0 = loadB(pB + (wc * 64 +  0) * GT_K, GT_K, lane);
        v16bf b1 = loadB(pB + (wc * 64 + 16) * GT_K, GT_K, lane);
        v16bf b2 = loadB(pB + (wc * 64 + 32) * GT_K, GT_K, lane);
        v16bf b3 = loadB(pB + (wc * 64 + 48) * GT_K, GT_K, lane);
        acc[0][0] = wmma_bf16(a0, b0, acc[0][0]);
        acc[1][0] = wmma_bf16(a1, b0, acc[1][0]);
        acc[0][1] = wmma_bf16(a0, b1, acc[0][1]);
        acc[1][1] = wmma_bf16(a1, b1, acc[1][1]);
        acc[0][2] = wmma_bf16(a0, b2, acc[0][2]);
        acc[1][2] = wmma_bf16(a1, b2, acc[1][2]);
        acc[0][3] = wmma_bf16(a0, b3, acc[0][3]);
        acc[1][3] = wmma_bf16(a1, b3, acc[1][3]);
    };

    const int ksteps = Kd / GT_K;
    issue_panels(0, 0);                         // prologue: step 0 in flight
    for (int s = 0; s < ksteps - 1; ++s) {      // branch-free hot loop
        const int buf = s & 1;
        issue_panels(buf ^ 1, (s + 1) * GT_K);  // keep next panel in flight
        asm volatile("s_wait_asynccnt 0x4" ::: "memory");  // retire step-s DMAs
        __syncthreads();                        // all waves' panel-s DMAs visible
        compute_step(buf);
        __syncthreads();                        // buf reusable for step s+2 DMAs
    }
    asm volatile("s_wait_asynccnt 0x0" ::: "memory");      // peeled last step
    __syncthreads();
    compute_step((ksteps - 1) & 1);

    // epilogue: C(m,n) layout -> lane n = lane&15, rows (lane>>4)*8 + r
    const int cn    = lane & 15;
    const int rbase = (lane >> 4) << 3;
#pragma unroll
    for (int i = 0; i < 2; ++i) {
#pragma unroll
        for (int j = 0; j < 4; ++j) {
            const int n = nb + wc * 64 + j * 16 + cn;
            const float bv = bias[n];
#pragma unroll
            for (int r = 0; r < 8; ++r) {
                const int m = mb + wr * 32 + i * 16 + rbase + r;
                const float val = acc[i][j][r] + bv;
                if (mode == 0) {
                    Cout[(size_t)m * N + n] = val;
                } else {
                    const int g  = n >> 10;            // 0:q 1:k 2:v
                    const int hd = n & (DM - 1);
                    const int h  = hd >> 6;
                    const int d  = hd & (HD - 1);
                    const int bb = m >> 11;            // m / SEQ
                    const int sq = m & (SEQ - 1);
                    const int bh = bb * NH + h;
                    if (g == 0)
                        Qs[((size_t)bh * SEQ + sq) * HD + d] = f2bf(val * 0.125f);
                    else if (g == 1)
                        Ks[((size_t)bh * SEQ + sq) * HD + d] = f2bf(val);
                    else
                        Vt[((size_t)bh * HD + d) * SEQ + sq] = f2bf(val);
                }
            }
        }
    }
}

// ---------------- attention: one (b,h,16-query-tile) per workgroup ----------------
// 8 waves. LDS: 128KB f32 scores [16][2048] + 64KB bf16 P [16][2048] (192KB of
// the WGP's 320KB). Exact softmax, no flash rescaling needed.
__global__ __launch_bounds__(256)
void attention_kernel(const unsigned short* __restrict__ Qs,
                      const unsigned short* __restrict__ Ks,
                      const unsigned short* __restrict__ Vt,
                      const unsigned char* __restrict__ mask,
                      unsigned short* __restrict__ attnout) {
    extern __shared__ char smem[];
    float*          sS = (float*)smem;                                // [16][SEQ]
    unsigned short* sP = (unsigned short*)(smem + 16 * SEQ * 4);      // [16][SEQ]

    const int tid  = threadIdx.x;
    const int lane = tid & 31;
    const int wave = tid >> 5;
    const int qt   = blockIdx.x & 127;       // query tile (128 per head)
    const int bh   = blockIdx.x >> 7;        // 0..31
    const int b    = bh >> 4;
    const int h    = bh & (NH - 1);
    const int cn    = lane & 15;
    const int rbase = (lane >> 4) << 3;

    // phase 1: S = (Q * scale) @ K^T  -- waves split the 128 key tiles
    const unsigned short* qptr = Qs + ((size_t)bh * SEQ + qt * 16) * HD;
    v16bf aq0 = loadA(qptr,      HD, lane);   // K-dim 0..31
    v16bf aq1 = loadA(qptr + 32, HD, lane);   // K-dim 32..63

    for (int kt = wave; kt < SEQ / 16; kt += 8) {
        const unsigned short* kptr = Ks + ((size_t)bh * SEQ + kt * 16) * HD;
        v16bf b0 = loadB(kptr,      HD, lane);
        v16bf b1 = loadB(kptr + 32, HD, lane);
        v8f acc = (v8f){};
        acc = wmma_bf16(aq0, b0, acc);
        acc = wmma_bf16(aq1, b1, acc);
        const int key = kt * 16 + cn;
        const float mbias = mask[b * SEQ + key] ? -1e30f : 0.0f;
#pragma unroll
        for (int r = 0; r < 8; ++r)
            sS[(rbase + r) * SEQ + key] = acc[r] + mbias;
    }
    __syncthreads();

    // phase 2: exact softmax; 16 lanes per row, shuffle reductions (wave32-safe)
    {
        const int row = tid >> 4;
        const int i   = tid & 15;
        float* srow = sS + row * SEQ;
        float mx = -1e30f;
        for (int k = i; k < SEQ; k += 16) mx = fmaxf(mx, srow[k]);
        for (int off = 8; off; off >>= 1) mx = fmaxf(mx, __shfl_xor(mx, off, 16));
        float sum = 0.0f;
        for (int k = i; k < SEQ; k += 16) {
            float e = __expf(srow[k] - mx);
            srow[k] = e;
            sum += e;
        }
        for (int off = 8; off; off >>= 1) sum += __shfl_xor(sum, off, 16);
        const float inv = 1.0f / sum;
        unsigned short* prow = sP + row * SEQ;
        for (int k = i; k < SEQ; k += 16) prow[k] = f2bf(srow[k] * inv);
    }
    __syncthreads();

    // phase 3: O = P @ V ; all 8 waves: wave = (dim block 0..3) x (key half 0..1).
    // Each wave contracts 1024 keys (32 WMMAs); halves merged via 4KB LDS reduce.
    {
        const int nb = (wave & 3) * 16;          // 16 head dims
        const int kh = wave >> 2;                // key half
        const unsigned short* vbase = Vt + ((size_t)bh * HD + nb) * SEQ;
        v8f o = (v8f){};
        for (int kc = kh * (SEQ / 2); kc < (kh + 1) * (SEQ / 2); kc += 32) {
            v16bf a  = loadA(sP + kc,    SEQ, lane);   // P from LDS
            v16bf bv = loadB(vbase + kc, SEQ, lane);   // V^T from global
            o = wmma_bf16(a, bv, o);
        }
        float* red = sS;                         // reuse score buffer: [16 rows][64 dims]
        if (kh == 1) {
#pragma unroll
            for (int r = 0; r < 8; ++r)
                red[(rbase + r) * HD + nb + cn] = o[r];   // conflict-free: lanes consecutive
        }
        __syncthreads();
        if (kh == 0) {
#pragma unroll
            for (int r = 0; r < 8; ++r) {
                const float v2 = red[(rbase + r) * HD + nb + cn];
                const int s = qt * 16 + rbase + r;
                attnout[((size_t)(b * SEQ + s)) * DM + h * HD + nb + cn] = f2bf(o[r] + v2);
            }
        }
    }
}

// ---------------- host side ----------------
extern "C" void kernel_launch(void* const* d_in, const int* in_sizes, int n_in,
                              void* d_out, int out_size, void* d_ws, size_t ws_size,
                              hipStream_t stream) {
    const float*         x    = (const float*)d_in[0];
    const unsigned char* mask = (const unsigned char*)d_in[1];
    const float*         wqkv = (const float*)d_in[2];
    const float*         bqkv = (const float*)d_in[3];
    const float*         wout = (const float*)d_in[4];
    const float*         bout = (const float*)d_in[5];
    float*               out  = (float*)d_out;

    // workspace layout (all 256B aligned)
    char* ws = (char*)d_ws;
    size_t off = 0;
    auto carve = [&](size_t bytes) {
        char* p = ws + off;
        off += (bytes + 255) & ~(size_t)255;
        return p;
    };
    unsigned short* xb    = (unsigned short*)carve((size_t)MROWS * DM * 2);   // 8 MB
    unsigned short* wqkvb = (unsigned short*)carve((size_t)NQKV * DM * 2);    // 6 MB
    unsigned short* woutb = (unsigned short*)carve((size_t)DM * DM * 2);      // 2 MB
    unsigned short* qs    = (unsigned short*)carve((size_t)MROWS * DM * 2);   // 8 MB
    unsigned short* ks    = (unsigned short*)carve((size_t)MROWS * DM * 2);   // 8 MB
    unsigned short* vt    = (unsigned short*)carve((size_t)MROWS * DM * 2);   // 8 MB
    unsigned short* attn  = (unsigned short*)carve((size_t)MROWS * DM * 2);   // 8 MB
    (void)ws_size; (void)in_sizes; (void)n_in; (void)out_size;

    // 1) convert inputs to bf16
    {
        int n1 = MROWS * DM;
        cvt_bf16_kernel<<<(n1 + 255) / 256, 256, 0, stream>>>(x, xb, n1);
        int n2 = NQKV * DM;
        cvt_bf16_kernel<<<(n2 + 255) / 256, 256, 0, stream>>>(wqkv, wqkvb, n2);
        int n3 = DM * DM;
        cvt_bf16_kernel<<<(n3 + 255) / 256, 256, 0, stream>>>(wout, woutb, n3);
    }

    // 2) QKV projection, fused scatter into Q/K (b,h,s,d) and V^T (b,h,d,s)
    {
        dim3 grid(NQKV / GT_N, MROWS / GT_M);
        gemm_bf16_lds_kernel<<<grid, 256, 0, stream>>>(xb, wqkvb, bqkv,
                                                       nullptr, qs, ks, vt,
                                                       NQKV, DM, /*mode=*/1);
    }

    // 3) attention: one workgroup per (b, h, 16-query tile); 192KB LDS
    {
        dim3 grid(BATCH * NH * (SEQ / 16));
        size_t shmem = (size_t)16 * SEQ * 4 + (size_t)16 * SEQ * 2;   // 196608 B
        attention_kernel<<<grid, 256, shmem, stream>>>(qs, ks, vt, mask, attn);
    }

    // 4) output projection -> f32 result
    {
        dim3 grid(DM / GT_N, MROWS / GT_M);
        gemm_bf16_lds_kernel<<<grid, 256, 0, stream>>>(attn, woutb, bout,
                                                       out, nullptr, nullptr, nullptr,
                                                       DM, DM, /*mode=*/0);
    }
}
